// SelfAttention_3805341024469
// MI455X (gfx1250) — matrix operand order repaired and checked by reading, not compiled, not run on hardware
//
#include <hip/hip_runtime.h>
#include <hip/hip_bf16.h>
#include <math.h>

// ---------------------------------------------------------------------------
// Self-attention block for MI455X (gfx1250, wave32, WMMA).
// Matrix work on v_wmma_f32_16x16x32_f16; GEMM tiles double-buffered in LDS
// via GLOBAL_LOAD_ASYNC_TO_LDS_B128 (builtin confirmed present by probe).
// ---------------------------------------------------------------------------

#define DIM      2048
#define SEQ      2048
#define NBATCH   2
#define NHEADS   32
#define HEADD    64
#define ROWS     (NBATCH * SEQ)          // 4096 GEMM rows
#define NQKV     (3 * DIM)               // 6144
#define LN_EPS   1e-5f

typedef _Float16 v8h  __attribute__((ext_vector_type(8)));
typedef _Float16 v16h __attribute__((ext_vector_type(16)));
typedef float    v8f  __attribute__((ext_vector_type(8)));
typedef int      v4i  __attribute__((ext_vector_type(4)));

#define NEG_BIG  (-1.0e30f)              // finite "-inf": exp2(NEG_BIG - m) == 0
#define LOG2E    1.44269504088896f

// ----- CDNA5 async global->LDS path (builtin present on this toolchain) ----
#if defined(__has_builtin)
#  if __has_builtin(__builtin_amdgcn_global_load_async_to_lds_b128)
#    define HAVE_ASYNC_B128 1
#  endif
#endif

typedef v4i __attribute__((address_space(1)))* gbl_v4i_p;   // printed __device__*
typedef v4i __attribute__((address_space(3)))* lds_v4i_p;   // printed __shared__*

__device__ __forceinline__ void copy_b128(const _Float16* g, _Float16* l) {
#ifdef HAVE_ASYNC_B128
    // (AS1 int4* src, AS3 int4* dst, imm offset, imm cpol). Generic->AS casts
    // via integer reinterpret: generic LDS addr truncates to 32-bit LDS offset.
    __builtin_amdgcn_global_load_async_to_lds_b128(
        (gbl_v4i_p)(uintptr_t)g, (lds_v4i_p)(uint32_t)(uintptr_t)l, 0, 0);
#else
    *(v8h*)l = *(const v8h*)g;
#endif
}

#ifdef HAVE_ASYNC_B128
#  if __has_builtin(__builtin_amdgcn_s_wait_asynccnt)
#    define WAIT_ASYNC(n) __builtin_amdgcn_s_wait_asynccnt(n)
#  else
#    define WAIT_ASYNC(n) asm volatile("s_wait_asynccnt %0" :: "i"(n) : "memory")
#  endif
#else
#  define WAIT_ASYNC(n) do {} while (0)
#endif

// ---------------------------------------------------------------------------
// Fragment loaders per CDNA5 ISA 7.12.2 (16-bit, wave32).
// A (16x32, MxK): lanes 0-15 row=lane, K chunks [0..7],[16..23];
//                 lanes 16-31 row=lane-16, K chunks [8..15],[24..31].
// B (32x16, KxN): lanes 0-15 col=lane, K chunks [0..7],[16..23]; lanes 16-31 K chunks [8..15],[24..31].
//                 (B loaded from a K-contiguous / transposed buffer.)
// ---------------------------------------------------------------------------
__device__ __forceinline__ v16h load_frag_a(const _Float16* p, int stride, int lane) {
    const int r  = lane & 15;
    const int hi = lane >> 4;
    const _Float16* q = p + r * stride + hi * 8;
    v8h lo = *(const v8h*)(q);
    v8h hh = *(const v8h*)(q + 16);
    return __builtin_shufflevector(lo, hh, 0,1,2,3,4,5,6,7,8,9,10,11,12,13,14,15);
}

__device__ __forceinline__ v16h load_frag_b(const _Float16* p, int stride, int lane) {
    const int n  = lane & 15;
    const int hi = lane >> 4;
    const _Float16* q = p + n * stride + hi * 8;
    v8h lo = *(const v8h*)(q);
    v8h hh = *(const v8h*)(q + 16);
    return __builtin_shufflevector(lo, hh, 0,1,2,3,4,5,6,7,8,9,10,11,12,13,14,15);
}

__device__ __forceinline__ v8f wmma_f16(v16h a, v16h b, v8f c) {
    return __builtin_amdgcn_wmma_f32_16x16x32_f16(false, a, false, b, (short)0, c,
                                                  false, false);
}

// ---------------------------------------------------------------------------
// Shared WMMA GEMM core: C[128x128] tile, 8 waves (2M x 4N), K-step 32,
// double-buffered LDS staged by async copies. acc[4][2] per wave.
// As/Bs must each hold 2*128*40 halves.
// ---------------------------------------------------------------------------
__device__ __forceinline__ void gemm_core(const _Float16* __restrict__ A,
                                          const _Float16* __restrict__ Bt,
                                          _Float16* As, _Float16* Bs,
                                          int m0, int n0, int t, v8f acc[4][2]) {
    const int lane = t & 31;
    const int wv   = t >> 5;
    const int wm   = wv & 1;
    const int wn   = wv >> 1;
    const int lr   = t >> 1;            // 128 tile rows, 2 threads/row
    const int lc   = (t & 1) * 16;      // 16 halves per thread

    const _Float16* gA = A  + (size_t)(m0 + lr) * DIM + lc;
    const _Float16* gB = Bt + (size_t)(n0 + lr) * DIM + lc;

    auto issue = [&](int kk, int buf) {
        _Float16* dA = As + buf * (128 * 40) + lr * 40 + lc;
        _Float16* dB = Bs + buf * (128 * 40) + lr * 40 + lc;
        copy_b128(gA + kk,     dA);
        copy_b128(gA + kk + 8, dA + 8);
        copy_b128(gB + kk,     dB);
        copy_b128(gB + kk + 8, dB + 8);
    };

    issue(0, 0);                        // stage first K tile

    for (int kt = 0; kt < DIM; kt += 32) {
        const int cur = (kt >> 5) & 1;
        if (kt + 32 < DIM) {
            issue(kt + 32, cur ^ 1);    // prefetch next tile into other buffer
            WAIT_ASYNC(4);              // previous 4 async ops complete
        } else {
            WAIT_ASYNC(0);
        }
        __syncthreads();                // tile `cur` visible to all waves

        const _Float16* ab = As + cur * (128 * 40);
        const _Float16* bb = Bs + cur * (128 * 40);
        v16h af[4], bf[2];
        #pragma unroll
        for (int mi = 0; mi < 4; mi++)
            af[mi] = load_frag_a(ab + (wm * 64 + mi * 16) * 40, 40, lane);
        #pragma unroll
        for (int ni = 0; ni < 2; ni++)
            bf[ni] = load_frag_b(bb + (wn * 32 + ni * 16) * 40, 40, lane);
        #pragma unroll
        for (int mi = 0; mi < 4; mi++)
            #pragma unroll
            for (int ni = 0; ni < 2; ni++)
                acc[mi][ni] = wmma_f16(af[mi], bf[ni], acc[mi][ni]);
        __syncthreads();                // done reading `cur` before overwrite
    }
}

// ---------------------------------------------------------------------------
// 1) LayerNorm (fp32 stats) -> f16 activations.
// ---------------------------------------------------------------------------
__global__ __launch_bounds__(256)
void ln_kernel(const float* __restrict__ x, const float* __restrict__ w,
               const float* __restrict__ b, _Float16* __restrict__ h16) {
    __shared__ float ssum[256];
    __shared__ float ssq[256];
    const int row = blockIdx.x;
    const int t   = threadIdx.x;
    const float* xr = x + (size_t)row * DIM;

    float s = 0.f, sq = 0.f;
    #pragma unroll
    for (int j = 0; j < DIM / 256; j++) {
        float v = xr[t + j * 256];
        s += v; sq += v * v;
    }
    ssum[t] = s; ssq[t] = sq;
    __syncthreads();
    for (int off = 128; off > 0; off >>= 1) {
        if (t < off) { ssum[t] += ssum[t + off]; ssq[t] += ssq[t + off]; }
        __syncthreads();
    }
    const float mu  = ssum[0] * (1.f / DIM);
    const float var = ssq[0] * (1.f / DIM) - mu * mu;
    const float rs  = rsqrtf(var + LN_EPS);

    _Float16* hr = h16 + (size_t)row * DIM;
    #pragma unroll
    for (int j = 0; j < DIM / 256; j++) {
        int c = t + j * 256;
        float v = (xr[c] - mu) * rs * w[c] + b[c];
        hr[c] = (_Float16)v;
    }
}

// ---------------------------------------------------------------------------
// 2) Weight transpose + f32->f16 convert: in[K][N] -> out[N][K].
// ---------------------------------------------------------------------------
__global__ __launch_bounds__(256)
void wconv_kernel(const float* __restrict__ in, _Float16* __restrict__ out,
                  int K, int N) {
    size_t idx = (size_t)blockIdx.x * 256 + threadIdx.x;
    size_t total = (size_t)K * N;
    if (idx >= total) return;
    int k = (int)(idx / N);
    int n = (int)(idx % N);
    out[(size_t)n * K + k] = (_Float16)in[idx];
}

// ---------------------------------------------------------------------------
// 3) QKV GEMM + bias, scatter into q[n,h,s,64], k[n,h,s,64], vT[n,h,64,s].
// ---------------------------------------------------------------------------
__global__ __launch_bounds__(256)
void gemm_qkv_kernel(const _Float16* __restrict__ A, const _Float16* __restrict__ Bt,
                     const float* __restrict__ bias,
                     _Float16* __restrict__ qb, _Float16* __restrict__ kb,
                     _Float16* __restrict__ vtb) {
    __shared__ alignas(32) _Float16 As[2 * 128 * 40];
    __shared__ alignas(32) _Float16 Bs[2 * 128 * 40];

    const int t  = threadIdx.x;
    const int m0 = blockIdx.y * 128;
    const int n0 = blockIdx.x * 128;

    v8f acc[4][2];
    #pragma unroll
    for (int mi = 0; mi < 4; mi++)
        #pragma unroll
        for (int ni = 0; ni < 2; ni++)
            #pragma unroll
            for (int i = 0; i < 8; i++) acc[mi][ni][i] = 0.f;

    gemm_core(A, Bt, As, Bs, m0, n0, t, acc);

    const int lane = t & 31, wv = t >> 5, wm = wv & 1, wn = wv >> 1;
    const int hi = lane >> 4, ln = lane & 15;
    #pragma unroll
    for (int mi = 0; mi < 4; mi++) {
        #pragma unroll
        for (int ni = 0; ni < 2; ni++) {
            #pragma unroll
            for (int i = 0; i < 8; i++) {
                int m = m0 + wm * 64 + mi * 16 + i + hi * 8;
                int n = n0 + wn * 32 + ni * 16 + ln;
                float v = acc[mi][ni][i] + bias[n];
                _Float16 hv = (_Float16)v;
                int tt = n >> 11;            // 0:q 1:k 2:v
                int hh = (n >> 6) & 31;      // head
                int d  = n & 63;             // dim in head
                int bb = m >> 11;            // batch
                int sp = m & 2047;           // position
                size_t nh = (size_t)bb * NHEADS + hh;
                if (tt == 0)      qb[(nh * SEQ + sp) * HEADD + d] = hv;
                else if (tt == 1) kb[(nh * SEQ + sp) * HEADD + d] = hv;
                else              vtb[(nh * HEADD + d) * SEQ + sp] = hv;
            }
        }
    }
}

// ---------------------------------------------------------------------------
// 4) In-place RoPE on q and k: pairs (d, d+16), d in [0,16).
// ---------------------------------------------------------------------------
__global__ __launch_bounds__(256)
void rope_kernel(_Float16* __restrict__ q, _Float16* __restrict__ k) {
    int idx = blockIdx.x * 256 + threadIdx.x;
    int d  = idx & 15;
    int sp = (idx >> 4) & (SEQ - 1);
    int nh = idx >> 15;
    if (nh >= NBATCH * NHEADS) return;

    // 10000^(-d/16) = exp2(-d * log2(10000)/16)
    float freq = exp2f(-(float)d * (13.2877123795494f / 16.f));
    float ang  = (float)sp * freq;
    float c = cosf(ang), s = sinf(ang);

    size_t base = ((size_t)nh * SEQ + sp) * HEADD;
    {
        float x1 = (float)q[base + d], x2 = (float)q[base + d + 16];
        q[base + d]      = (_Float16)(x1 * c - x2 * s);
        q[base + d + 16] = (_Float16)(x2 * c + x1 * s);
    }
    {
        float x1 = (float)k[base + d], x2 = (float)k[base + d + 16];
        k[base + d]      = (_Float16)(x1 * c - x2 * s);
        k[base + d + 16] = (_Float16)(x2 * c + x1 * s);
    }
}

// ---------------------------------------------------------------------------
// 5) Causal flash attention; one wave per 16-query tile, 32-key blocks.
//    Softmax kept in the log2 domain: exp2f -> native v_exp_f32.
//    Since k0 % 32 == 0 <= qs (qs % 16 == 0), no row is ever fully masked,
//    so a finite NEG_BIG sentinel needs no NaN guards.
// ---------------------------------------------------------------------------
__global__ __launch_bounds__(128)
void attn_kernel(const _Float16* __restrict__ q, const _Float16* __restrict__ k,
                 const _Float16* __restrict__ vt, _Float16* __restrict__ o16) {
    __shared__ alignas(32) _Float16 Pl[4 * 16 * 32];

    const int lane = threadIdx.x & 31;
    const int wv   = threadIdx.x >> 5;
    const int hi   = lane >> 4, ln = lane & 15;

    const int qtile = blockIdx.x * 4 + wv;
    const int qs    = qtile * 16;
    const int nh    = blockIdx.y;
    const int bb    = nh >> 5;
    const int hd    = nh & 31;

    const _Float16* qbase = q  + (size_t)nh * SEQ * HEADD;
    const _Float16* kbase = k  + (size_t)nh * SEQ * HEADD;
    const _Float16* vbase = vt + (size_t)nh * HEADD * SEQ;
    _Float16* pw = &Pl[wv * 16 * 32];

    v16h aq[2];
    #pragma unroll
    for (int kk = 0; kk < 2; kk++)
        aq[kk] = load_frag_a(qbase + (size_t)qs * HEADD + kk * 32, HEADD, lane);

    v8f o[4];
    float rm[8], rl[8];
    #pragma unroll
    for (int dt = 0; dt < 4; dt++)
        #pragma unroll
        for (int i = 0; i < 8; i++) o[dt][i] = 0.f;
    #pragma unroll
    for (int i = 0; i < 8; i++) { rm[i] = NEG_BIG; rl[i] = 0.f; }

    const float sc2 = 0.125f * LOG2E;   // 1/sqrt(64) folded with log2(e)

    for (int k0 = 0; k0 <= qs + 15; k0 += 32) {
        if (k0 + 32 <= qs) {            // prefetch next K/V block
            __builtin_prefetch(kbase + (size_t)(k0 + 32) * HEADD, 0, 0);
            __builtin_prefetch(vbase + (size_t)(k0 + 32), 0, 0);
        }

        // ---- S = Q K^T (16 x 32 scores) ----
        v8f s0, s1;
        #pragma unroll
        for (int i = 0; i < 8; i++) { s0[i] = 0.f; s1[i] = 0.f; }
        #pragma unroll
        for (int kk = 0; kk < 2; kk++) {
            v16h b0 = load_frag_b(kbase + (size_t)(k0)      * HEADD + kk * 32, HEADD, lane);
            v16h b1 = load_frag_b(kbase + (size_t)(k0 + 16) * HEADD + kk * 32, HEADD, lane);
            s0 = wmma_f16(aq[kk], b0, s0);
            s1 = wmma_f16(aq[kk], b1, s1);
        }

        // ---- online softmax (log2 domain), rows i / i+8 per lane half ----
        float alpha[8];
        #pragma unroll
        for (int i = 0; i < 8; i++) {
            int grow = qs + i + hi * 8;
            int key0 = k0 + ln;
            float e0 = s0[i] * sc2; if (key0 > grow)      e0 = NEG_BIG;
            float e1 = s1[i] * sc2; if (key0 + 16 > grow) e1 = NEG_BIG;

            float tm = fmaxf(e0, e1);
            #pragma unroll
            for (int off = 1; off < 16; off <<= 1)
                tm = fmaxf(tm, __shfl_xor(tm, off, 32));
            float mn = fmaxf(rm[i], tm);

            float al = exp2f(rm[i] - mn);   // 0 on first block (underflow)
            float p0 = exp2f(e0 - mn);      // 0 for masked entries
            float p1 = exp2f(e1 - mn);

            float ps = p0 + p1;
            #pragma unroll
            for (int off = 1; off < 16; off <<= 1)
                ps += __shfl_xor(ps, off, 32);

            rl[i] = rl[i] * al + ps;
            rm[i] = mn;
            alpha[i] = al;

            _Float16* pr = pw + (i + hi * 8) * 32;
            pr[ln]      = (_Float16)p0;
            pr[16 + ln] = (_Float16)p1;
        }

        asm volatile("s_wait_dscnt 0" ::: "memory");
        __builtin_amdgcn_wave_barrier();

        // ---- O = alpha*O + P V ----
        v16h ap = load_frag_a(pw, 32, lane);
        #pragma unroll
        for (int dt = 0; dt < 4; dt++) {
            v16h bv = load_frag_b(vbase + (size_t)(dt * 16) * SEQ + k0, SEQ, lane);
            #pragma unroll
            for (int i = 0; i < 8; i++) o[dt][i] *= alpha[i];
            o[dt] = wmma_f16(ap, bv, o[dt]);
        }
        __builtin_amdgcn_wave_barrier();
    }

    float inv[8];
    #pragma unroll
    for (int i = 0; i < 8; i++) inv[i] = (rl[i] > 0.f) ? 1.f / rl[i] : 0.f;
    #pragma unroll
    for (int dt = 0; dt < 4; dt++) {
        #pragma unroll
        for (int i = 0; i < 8; i++) {
            int grow = bb * SEQ + qs + i + hi * 8;
            int gcol = hd * HEADD + dt * 16 + ln;
            o16[(size_t)grow * DIM + gcol] = (_Float16)(o[dt][i] * inv[i]);
        }
    }
}

// ---------------------------------------------------------------------------
// 6) Output GEMM: o16 @ woutT + b_out + skip -> f32.
// ---------------------------------------------------------------------------
__global__ __launch_bounds__(256)
void gemm_out_kernel(const _Float16* __restrict__ A, const _Float16* __restrict__ Bt,
                     const float* __restrict__ bias, const float* __restrict__ skip,
                     float* __restrict__ out) {
    __shared__ alignas(32) _Float16 As[2 * 128 * 40];
    __shared__ alignas(32) _Float16 Bs[2 * 128 * 40];

    const int t  = threadIdx.x;
    const int m0 = blockIdx.y * 128;
    const int n0 = blockIdx.x * 128;

    v8f acc[4][2];
    #pragma unroll
    for (int mi = 0; mi < 4; mi++)
        #pragma unroll
        for (int ni = 0; ni < 2; ni++)
            #pragma unroll
            for (int i = 0; i < 8; i++) acc[mi][ni][i] = 0.f;

    gemm_core(A, Bt, As, Bs, m0, n0, t, acc);

    const int lane = t & 31, wv = t >> 5, wm = wv & 1, wn = wv >> 1;
    const int hi = lane >> 4, ln = lane & 15;
    #pragma unroll
    for (int mi = 0; mi < 4; mi++) {
        #pragma unroll
        for (int ni = 0; ni < 2; ni++) {
            #pragma unroll
            for (int i = 0; i < 8; i++) {
                int m = m0 + wm * 64 + mi * 16 + i + hi * 8;
                int n = n0 + wn * 32 + ni * 16 + ln;
                size_t gi = (size_t)m * DIM + n;
                out[gi] = acc[mi][ni][i] + bias[n] + skip[gi];
            }
        }
    }
}

// ---------------------------------------------------------------------------
// Launcher. Workspace (~112 MB):
//   h16 @0 (16MB), wqkvT @16MB (24MB), woutT @40MB (8MB),
//   q @48MB, k @64MB, vT @80MB, o16 @96MB (16MB each).
// ---------------------------------------------------------------------------
extern "C" void kernel_launch(void* const* d_in, const int* in_sizes, int n_in,
                              void* d_out, int out_size, void* d_ws, size_t ws_size,
                              hipStream_t stream) {
    const float* x     = (const float*)d_in[0];
    const float* ln_w  = (const float*)d_in[1];
    const float* ln_b  = (const float*)d_in[2];
    const float* w_qkv = (const float*)d_in[3];
    const float* b_qkv = (const float*)d_in[4];
    const float* w_out = (const float*)d_in[5];
    const float* b_out = (const float*)d_in[6];
    float* out = (float*)d_out;

    char* ws = (char*)d_ws;
    _Float16* h16   = (_Float16*)(ws);
    _Float16* wqkvT = (_Float16*)(ws + (size_t)16 * 1024 * 1024);
    _Float16* woutT = (_Float16*)(ws + (size_t)40 * 1024 * 1024);
    _Float16* qb    = (_Float16*)(ws + (size_t)48 * 1024 * 1024);
    _Float16* kb    = (_Float16*)(ws + (size_t)64 * 1024 * 1024);
    _Float16* vtb   = (_Float16*)(ws + (size_t)80 * 1024 * 1024);
    _Float16* o16   = (_Float16*)(ws + (size_t)96 * 1024 * 1024);

    ln_kernel<<<ROWS, 256, 0, stream>>>(x, ln_w, ln_b, h16);

    {
        size_t tq = (size_t)DIM * NQKV;
        wconv_kernel<<<(unsigned)((tq + 255) / 256), 256, 0, stream>>>(w_qkv, wqkvT, DIM, NQKV);
        size_t to = (size_t)DIM * DIM;
        wconv_kernel<<<(unsigned)((to + 255) / 256), 256, 0, stream>>>(w_out, woutT, DIM, DIM);
    }

    gemm_qkv_kernel<<<dim3(NQKV / 128, ROWS / 128), 256, 0, stream>>>(
        h16, wqkvT, b_qkv, qb, kb, vtb);

    {
        int total = NBATCH * NHEADS * SEQ * 16;
        rope_kernel<<<total / 256, 256, 0, stream>>>(qb, kb);
    }

    attn_kernel<<<dim3(SEQ / 64, NBATCH * NHEADS), 128, 0, stream>>>(qb, kb, vtb, o16);

    gemm_out_kernel<<<dim3(DIM / 128, ROWS / 128), 256, 0, stream>>>(
        o16, woutT, b_out, x, out);
}
